// JAISAttention_34359738680
// MI455X (gfx1250) — matrix operand order repaired and checked by reading, not compiled, not run on hardware
//
#include <hip/hip_runtime.h>
#include <stdint.h>

typedef unsigned short u16;
typedef __attribute__((ext_vector_type(16))) __bf16 v16bf;
typedef __attribute__((ext_vector_type(8)))  float  v8f;

#define S_LEN 2048
#define BATCH 2
#define HID   4096
#define NHEAD 32
#define HDIM  128
#define MTOT  (BATCH * S_LEN)   // 4096 rows total
#define NQKV  (3 * HID)         // 12288
#define ATT_SCALE (1.0f / 128.0f)

union Frag {
  v16bf v;
  uint4 q[2];
};

__device__ __forceinline__ u16 f2bf(float f) {
  uint32_t u = __float_as_uint(f);
  u += 0x7FFFu + ((u >> 16) & 1u);   // round to nearest even
  return (u16)(u >> 16);
}

__device__ __forceinline__ uint4 ld16(const u16* p) {
  uint4 t; __builtin_memcpy(&t, p, sizeof(uint4)); return t;
}

// Low 32 bits of a flat pointer into LDS == wave-relative LDS byte address
// (flat->LDS aperture mapping truncates to addr[31:0]).
__device__ __forceinline__ unsigned lds_off(const void* p) {
  return (unsigned)(uintptr_t)p;
}

// Async copy global -> LDS (tracked on ASYNCcnt). The instruction offset is
// added to BOTH the LDS and the memory address.
__device__ __forceinline__ void async_cp64(unsigned lds, const u16* gp) {
  asm volatile("global_load_async_to_lds_b128 %0, %1, off\n\t"
               "global_load_async_to_lds_b128 %0, %1, off offset:16\n\t"
               "global_load_async_to_lds_b128 %0, %1, off offset:32\n\t"
               "global_load_async_to_lds_b128 %0, %1, off offset:48"
               :: "v"(lds), "v"(gp) : "memory");
}
__device__ __forceinline__ void wait_async0() {
  asm volatile("s_wait_asynccnt 0x0" ::: "memory");
}

// ---------------------------------------------------------------------------
// fp32 -> bf16 conversion (grid-stride, 4 elements per thread-step)
// ---------------------------------------------------------------------------
__global__ void cvt_f32_to_bf16_v4(const float4* __restrict__ in,
                                   uint2* __restrict__ out, long long n4) {
  long long i = (long long)blockIdx.x * blockDim.x + threadIdx.x;
  long long stride = (long long)gridDim.x * blockDim.x;
  for (; i < n4; i += stride) {
    float4 f = in[i];
    union { uint2 u; u16 s[4]; } r;
    r.s[0] = f2bf(f.x); r.s[1] = f2bf(f.y);
    r.s[2] = f2bf(f.z); r.s[3] = f2bf(f.w);
    out[i] = r.u;
  }
}

// ---------------------------------------------------------------------------
// bf16 WMMA GEMM: C[M,N] = A[M,K] @ B[K,N] + bias[N]
// block tile 128x128, 128 threads = 4 waves, each wave 64x64 (4x4 WMMA tiles)
// Double-buffered LDS; A tile staged via async global->LDS copies.
// ---------------------------------------------------------------------------
template <int OUT_BF16>
__global__ __launch_bounds__(128)
void gemm_bf16_wmma(const u16* __restrict__ A, const u16* __restrict__ B,
                    const float* __restrict__ bias, void* __restrict__ Cout,
                    int M, int N, int K) {
  __shared__ __align__(16) u16 As[2][128 * 32];   // [m][k]
  __shared__ __align__(16) u16 Bs[2][128 * 32];   // transposed: [n][k]

  const int tid  = threadIdx.x;
  const int lane = tid & 31;
  const int wid  = tid >> 5;        // 0..3
  const int h    = lane >> 4;       // half-wave group
  const int ln   = lane & 15;
  const int wm   = (wid & 1) * 64;  // wave m offset in block tile
  const int wn   = (wid >> 1) * 64; // wave n offset in block tile
  const int bM   = blockIdx.y * 128;
  const int bN   = blockIdx.x * 128;

  v8f acc[4][4] = {};

  const int arow  = tid;                 // 0..127 (full 32-half row per thread)
  const int brow  = tid >> 2;            // 0..31 (k row)
  const int bpart = (tid & 3) * 32;      // 0..96 (n chunk)
  const int nK    = K >> 5;

  union BV { uint4 q[4]; u16 hh[32]; };

  // ---- prologue: stage tile 0 ----
  async_cp64(lds_off(&As[0][arow * 32]), A + (size_t)(bM + arow) * K);
  {
    BV tb;
    const u16* gb = B + (size_t)brow * N + bN + bpart;
    tb.q[0] = ld16(gb);      tb.q[1] = ld16(gb + 8);
    tb.q[2] = ld16(gb + 16); tb.q[3] = ld16(gb + 24);
#pragma unroll
    for (int j = 0; j < 32; ++j)
      Bs[0][(bpart + j) * 32 + brow] = tb.hh[j];
  }
  wait_async0();
  __syncthreads();

  for (int i = 0; i < nK; ++i) {
    const int cur = i & 1;
    const int nxt = cur ^ 1;
    const bool more = (i + 1) < nK;

    // ---- prefetch tile i+1 ----
    BV nb;
    if (more) {
      const int k0n = (i + 1) << 5;
      async_cp64(lds_off(&As[nxt][arow * 32]),
                 A + (size_t)(bM + arow) * K + k0n);
      const u16* gb = B + (size_t)(k0n + brow) * N + bN + bpart;
      nb.q[0] = ld16(gb);      nb.q[1] = ld16(gb + 8);
      nb.q[2] = ld16(gb + 16); nb.q[3] = ld16(gb + 24);
    }

    // ---- compute tile i: 4x4 WMMA tiles per wave ----
    Frag af[4], bf[4];
#pragma unroll
    for (int t = 0; t < 4; ++t) {
      const int m = wm + t * 16 + ln;
      af[t].q[0] = ld16(&As[cur][m * 32 + 8 * h]);        // k = 8h..8h+7
      af[t].q[1] = ld16(&As[cur][m * 32 + 16 + 8 * h]);   // k = 16+8h..
    }
#pragma unroll
    for (int u = 0; u < 4; ++u) {
      const int n = wn + u * 16 + ln;
      bf[u].q[0] = ld16(&Bs[cur][n * 32 + 8 * h]);        // k = 8h..8h+7
      bf[u].q[1] = ld16(&Bs[cur][n * 32 + 16 + 8 * h]);
    }
#pragma unroll
    for (int t = 0; t < 4; ++t)
#pragma unroll
      for (int u = 0; u < 4; ++u)
        acc[t][u] = __builtin_amdgcn_wmma_f32_16x16x32_bf16(
            false, af[t].v, false, bf[u].v, (short)0, acc[t][u], false, false);

    // ---- publish tile i+1 ----
    if (more) {
#pragma unroll
      for (int j = 0; j < 32; ++j)
        Bs[nxt][(bpart + j) * 32 + brow] = nb.hh[j];
      wait_async0();
    }
    __syncthreads();
  }

  // ---- epilogue: C layout = (row r+8h, col ln) ----
#pragma unroll
  for (int t = 0; t < 4; ++t)
#pragma unroll
    for (int u = 0; u < 4; ++u)
#pragma unroll
      for (int r = 0; r < 8; ++r) {
        const int row = bM + wm + t * 16 + r + 8 * h;
        const int col = bN + wn + u * 16 + ln;
        const float val = acc[t][u][r] + bias[col];
        if (OUT_BF16)
          ((u16*)Cout)[(size_t)row * N + col] = f2bf(val);
        else
          ((float*)Cout)[(size_t)row * N + col] = val;
      }
}

// ---------------------------------------------------------------------------
// Flash attention with ALiBi + causal mask, bf16 WMMA, fp32 accumulate.
// grid = (S/64, NHEAD, BATCH), block = 128 (4 waves, 16 q-rows per wave)
// Double-buffered LDS; K tile staged via async global->LDS copies.
// qkv: bf16 [MTOT][NQKV] (q | k | v sections); out: bf16 [MTOT][HID]
// ---------------------------------------------------------------------------
__global__ __launch_bounds__(128)
void jais_attn_wmma(const u16* __restrict__ qkv, u16* __restrict__ out) {
  __shared__ __align__(16) u16 Ks[2][32 * 128];    // [key][d]
  __shared__ __align__(16) u16 Vs[2][128 * 32];    // transposed: [d][key]
  __shared__ __align__(16) u16 Ps[4 * 16 * 32];    // per-wave P staging

  const int tid  = threadIdx.x;
  const int lane = tid & 31;
  const int wid  = tid >> 5;
  const int h    = lane >> 4;
  const int ln   = lane & 15;
  const int qblk = blockIdx.x;
  const int hh   = blockIdx.y;
  const int b    = blockIdx.z;
  const int q0   = qblk * 64 + wid * 16;
  const size_t rowbase = (size_t)b * S_LEN;
  const float slope = exp2f(-0.25f * (float)(hh + 1));  // 32 heads, pow2

  // Q fragments for this wave's 16-row tile (A layout: lane row = ln)
  Frag qa[4];
  {
    const u16* qp = qkv + (rowbase + q0 + ln) * NQKV + hh * HDIM;
#pragma unroll
    for (int c = 0; c < 4; ++c) {
      qa[c].q[0] = ld16(qp + c * 32 + 8 * h);
      qa[c].q[1] = ld16(qp + c * 32 + 16 + 8 * h);
    }
  }

  v8f o[8] = {};
  float mrun[8], lrun[8], sqg[8];
  int qgi[8];
#pragma unroll
  for (int r = 0; r < 8; ++r) {
    mrun[r] = -1.0e30f; lrun[r] = 0.0f;
    qgi[r] = q0 + r + 8 * h;                 // this lane's row index
    sqg[r] = slope * (float)qgi[r];          // hoisted slope*q
  }
  const float slope16 = slope * 16.0f;

  const int key_t = tid >> 2;          // 0..31: key within tile
  const int dpart = (tid & 3) * 32;    // 0..96: d chunk
  const int nkb = qblk * 2 + 2;        // key blocks of 32 up to causal limit

  union VV { uint4 q[4]; u16 hh16[32]; };

  // ---- prologue: stage key-block 0 ----
  async_cp64(lds_off(&Ks[0][key_t * 128 + dpart]),
             qkv + (rowbase + key_t) * NQKV + HID + hh * HDIM + dpart);
  {
    VV vr;
    const u16* vp = qkv + (rowbase + key_t) * NQKV + 2 * HID + hh * HDIM + dpart;
    vr.q[0] = ld16(vp);      vr.q[1] = ld16(vp + 8);
    vr.q[2] = ld16(vp + 16); vr.q[3] = ld16(vp + 24);
#pragma unroll
    for (int j = 0; j < 32; ++j)
      Vs[0][(dpart + j) * 32 + key_t] = vr.hh16[j];
  }
  wait_async0();
  __syncthreads();

  for (int kb = 0; kb < nkb; ++kb) {
    const int cur = kb & 1;
    const int nxt = cur ^ 1;
    const bool more = (kb + 1) < nkb;

    // ---- prefetch key-block kb+1 ----
    VV vr;
    if (more) {
      const size_t krow = rowbase + (size_t)(kb + 1) * 32 + key_t;
      async_cp64(lds_off(&Ks[nxt][key_t * 128 + dpart]),
                 qkv + krow * NQKV + HID + hh * HDIM + dpart);
      const u16* vp = qkv + krow * NQKV + 2 * HID + hh * HDIM + dpart;
      vr.q[0] = ld16(vp);      vr.q[1] = ld16(vp + 8);
      vr.q[2] = ld16(vp + 16); vr.q[3] = ld16(vp + 24);
    }

    // ---- scores: two 16x16 key subtiles, D = 4 x K32 ----
    v8f sc[2] = {};
#pragma unroll
    for (int kt = 0; kt < 2; ++kt)
#pragma unroll
      for (int c = 0; c < 4; ++c) {
        Frag kf;
        kf.q[0] = ld16(&Ks[cur][(kt * 16 + ln) * 128 + c * 32 + 8 * h]);
        kf.q[1] = ld16(&Ks[cur][(kt * 16 + ln) * 128 + c * 32 + 16 + 8 * h]);
        sc[kt] = __builtin_amdgcn_wmma_f32_16x16x32_bf16(
            false, qa[c].v, false, kf.v, (short)0, sc[kt], false, false);
      }

    // ---- online softmax (row = r + 8h, col = key) ----
    const int key0 = kb * 32 + ln;                 // this lane's key column
    const float sk0 = slope * (float)key0;         // hoisted slope*key
#pragma unroll
    for (int r = 0; r < 8; ++r) {
      float s0 = sc[0][r] * ATT_SCALE + (sk0 - sqg[r]);
      float s1 = sc[1][r] * ATT_SCALE + (sk0 + slope16 - sqg[r]);
      if (key0 > qgi[r])      s0 = -1.0e30f;
      if (key0 + 16 > qgi[r]) s1 = -1.0e30f;

      float mx = fmaxf(s0, s1);
#pragma unroll
      for (int msk = 1; msk < 16; msk <<= 1)
        mx = fmaxf(mx, __shfl_xor(mx, msk, 16));
      const float mnew = fmaxf(mrun[r], mx);
      const float alpha = __expf(mrun[r] - mnew);
      const float p0 = __expf(s0 - mnew);
      const float p1 = __expf(s1 - mnew);
      float rs = p0 + p1;
#pragma unroll
      for (int msk = 1; msk < 16; msk <<= 1)
        rs += __shfl_xor(rs, msk, 16);
      lrun[r] = lrun[r] * alpha + rs;
      mrun[r] = mnew;
#pragma unroll
      for (int u = 0; u < 8; ++u)
        o[u][r] *= alpha;
      // stage P (C layout -> LDS bf16 tile [16][32])
      Ps[wid * 512 + (r + 8 * h) * 32 + ln]      = f2bf(p0);
      Ps[wid * 512 + (r + 8 * h) * 32 + 16 + ln] = f2bf(p1);
    }

    // ---- P @ V: A = P[16x32], B = V[32x16] per d-tile ----
    Frag pa;
    pa.q[0] = ld16(&Ps[wid * 512 + ln * 32 + 8 * h]);
    pa.q[1] = ld16(&Ps[wid * 512 + ln * 32 + 16 + 8 * h]);
#pragma unroll
    for (int u = 0; u < 8; ++u) {
      Frag vf;
      vf.q[0] = ld16(&Vs[cur][(u * 16 + ln) * 32 + 8 * h]);
      vf.q[1] = ld16(&Vs[cur][(u * 16 + ln) * 32 + 16 + 8 * h]);
      o[u] = __builtin_amdgcn_wmma_f32_16x16x32_bf16(
          false, pa.v, false, vf.v, (short)0, o[u], false, false);
    }

    // ---- publish key-block kb+1 ----
    if (more) {
#pragma unroll
      for (int j = 0; j < 32; ++j)
        Vs[nxt][(dpart + j) * 32 + key_t] = vr.hh16[j];
      wait_async0();
    }
    __syncthreads();
  }

  // ---- epilogue: out[b,s][h*128+d] as bf16 for the projection GEMM ----
#pragma unroll
  for (int u = 0; u < 8; ++u)
#pragma unroll
    for (int r = 0; r < 8; ++r) {
      const int qg = q0 + r + 8 * h;
      const float val = o[u][r] / lrun[r];
      out[(rowbase + qg) * HID + hh * HDIM + u * 16 + ln] = f2bf(val);
    }
}

// ---------------------------------------------------------------------------
// Launch
// ---------------------------------------------------------------------------
extern "C" void kernel_launch(void* const* d_in, const int* in_sizes, int n_in,
                              void* d_out, int out_size, void* d_ws, size_t ws_size,
                              hipStream_t stream) {
  const float* hs     = (const float*)d_in[0];  // [B,S,HID]
  const float* w_attn = (const float*)d_in[1];  // [HID, 3*HID]
  const float* b_attn = (const float*)d_in[2];  // [3*HID]
  const float* w_proj = (const float*)d_in[3];  // [HID, HID]
  const float* b_proj = (const float*)d_in[4];  // [HID]
  float* out = (float*)d_out;

  // workspace layout (bf16 buffers)
  u16* X   = (u16*)d_ws;                          // [MTOT][HID]      32 MB
  u16* Wa  = X   + (size_t)MTOT * HID;            // [HID][NQKV]      96 MB
  u16* Wp  = Wa  + (size_t)HID * NQKV;            // [HID][HID]       32 MB
  u16* QKV = Wp  + (size_t)HID * HID;             // [MTOT][NQKV]     96 MB
  u16* AO  = QKV + (size_t)MTOT * NQKV;           // [MTOT][HID]      32 MB

  const int cvtBlocks = 2048, cvtThreads = 256;
  cvt_f32_to_bf16_v4<<<cvtBlocks, cvtThreads, 0, stream>>>(
      (const float4*)hs, (uint2*)X, (long long)MTOT * HID / 4);
  cvt_f32_to_bf16_v4<<<cvtBlocks, cvtThreads, 0, stream>>>(
      (const float4*)w_attn, (uint2*)Wa, (long long)HID * NQKV / 4);
  cvt_f32_to_bf16_v4<<<cvtBlocks, cvtThreads, 0, stream>>>(
      (const float4*)w_proj, (uint2*)Wp, (long long)HID * HID / 4);

  // QKV projection: [4096,4096] @ [4096,12288] + b_attn -> bf16
  gemm_bf16_wmma<1><<<dim3(NQKV / 128, MTOT / 128), 128, 0, stream>>>(
      X, Wa, b_attn, QKV, MTOT, NQKV, HID);

  // Attention (flash, ALiBi, causal) -> bf16 [MTOT][HID]
  jais_attn_wmma<<<dim3(S_LEN / 64, NHEAD, BATCH), 128, 0, stream>>>(QKV, AO);

  // Output projection: [4096,4096] @ [4096,4096] + b_proj -> f32 d_out
  gemm_bf16_wmma<0><<<dim3(HID / 128, MTOT / 128), 128, 0, stream>>>(
      AO, Wp, b_proj, out, MTOT, HID, HID);
}
